// QCircuitBipartiteModel_16286515986488
// MI455X (gfx1250) — compile-verified, matrix operand order checked by reading, not run on hardware
//
#include <hip/hip_runtime.h>
#include <hip/hip_bf16.h>
#include <math.h>

#define NG 500000
#define NQ 100000
#define NE 1000000
#define HD 128
#define NL 4
#define NB 1024
#define EMB 16
#define GDIM 32

#define APAD 136   // padded LDS row stride in halves (272B = 17*16B): 16B-aligned rows, conflict-free frag loads

typedef __attribute__((ext_vector_type(16))) __bf16 v16bf;
typedef __attribute__((ext_vector_type(8)))  float  v8f;

// ---------- helpers ----------
__device__ __forceinline__ unsigned short f2bf(float f) {
    union { float f; unsigned u; } c; c.f = f;
    unsigned r = c.u + 0x7FFFu + ((c.u >> 16) & 1u);   // round-to-nearest-even
    return (unsigned short)(r >> 16);
}
union Frag16 { unsigned short u[16]; int4 q[2]; v16bf v; };

__device__ __forceinline__ float silu(float x) { return x / (1.0f + __expf(-x)); }

__device__ __forceinline__ void atomic_add_f32(float* p, float v) {
    __hip_atomic_fetch_add(p, v, __ATOMIC_RELAXED, __HIP_MEMORY_SCOPE_AGENT);
}

// A/X fragment from LDS tile (padded stride APAD): 2x ds_load_b128
// lanes 0-15: halves 0..7 -> K=kc+0..7, halves 8..15 -> K=kc+16..23
// lanes 16-31: halves 0..7 -> K=kc+8..15, halves 8..15 -> K=kc+24..31
__device__ __forceinline__ v16bf load_frag_lds(const unsigned short* S, int r0, int lane, int kc) {
    int r  = r0 + (lane & 15);
    int k0 = kc + ((lane & 16) ? 8 : 0);
    Frag16 t;
    t.q[0] = *(const int4*)&S[r * APAD + k0];
    t.q[1] = *(const int4*)&S[r * APAD + k0 + 16];
    return t.v;
}
// B fragment from globally pre-packed weights: per-lane contiguous 32B -> 2x global_load_b128
__device__ __forceinline__ v16bf load_frag_packed(const unsigned short* __restrict__ P,
                                                  int lane, int kcIdx, int nt) {
    const int4* p = (const int4*)(P + (((kcIdx << 3) + nt) * 32 + lane) * 16);
    Frag16 t;
    t.q[0] = p[0];
    t.q[1] = p[1];
    return t.v;
}

// ---------- one-time weight packing: fp32 [nmat][128][128] -> bf16 fragment-major ----------
__global__ void pack_w_kernel(const float* __restrict__ W, unsigned short* __restrict__ P, int nmat) {
    size_t t = (size_t)blockIdx.x * blockDim.x + threadIdx.x;
    if (t >= (size_t)nmat * 16384) return;
    int m   = (int)(t >> 14);
    int idx = (int)(t & 16383);
    int i   = idx & 15;
    int l   = (idx >> 4) & 31;
    int f   = idx >> 9;             // 0..31 : kcIdx*8 + nt
    int kcIdx = f >> 3, nt = f & 7;
    int k = kcIdx * 32 + ((l & 16) ? 8 : 0) + (i & 7) + ((i >> 3) << 4);
    int n = nt * 16 + (l & 15);
    P[t] = f2bf(W[(size_t)m * 16384 + k * HD + n]);
}

// ---------- input projections ----------
__global__ void gate_proj_kernel(const int* __restrict__ type, const int* __restrict__ arity,
                                 const int* __restrict__ isdir, const float* __restrict__ idxn,
                                 const float* __restrict__ emb, const float* __restrict__ W,
                                 const float* __restrict__ b, float* __restrict__ gx, int ng) {
    int g = blockIdx.x;
    if (g >= ng) return;
    __shared__ float f[EMB + 3];
    int tid = threadIdx.x;
    if (tid < EMB) f[tid] = emb[type[g] * EMB + tid];
    if (tid == EMB)     f[EMB]     = (float)arity[g];
    if (tid == EMB + 1) f[EMB + 1] = (float)isdir[g];
    if (tid == EMB + 2) f[EMB + 2] = idxn[g];
    __syncthreads();
    float acc = b[tid];
#pragma unroll
    for (int k = 0; k < EMB + 3; ++k) acc += f[k] * W[k * HD + tid];
    gx[(size_t)g * HD + tid] = acc;
}

__global__ void qubit_proj_kernel(const float* __restrict__ deg, const float* __restrict__ W,
                                  const float* __restrict__ b, float* __restrict__ qx, int nq) {
    size_t t = (size_t)blockIdx.x * blockDim.x + threadIdx.x;
    if (t >= (size_t)nq * HD) return;
    int q = (int)(t >> 7), h = (int)(t & 127);
    qx[t] = deg[q] * W[h] + b[h];
}

// ---------- edge mean aggregation ----------
__global__ void scatter_add_kernel(const float* __restrict__ feat, const int* __restrict__ src,
                                   const int* __restrict__ dst, float* __restrict__ agg,
                                   float* __restrict__ cnt, int ne) {
    size_t t = (size_t)blockIdx.x * blockDim.x + threadIdx.x;
    int e = (int)(t >> 5);
    if (e >= ne) return;
    int c = (int)(t & 31) * 4;
    int s = src[e], d = dst[e];
    const float* sp = feat + (size_t)s * HD + c;
    float*       dp = agg  + (size_t)d * HD + c;
#pragma unroll
    for (int i = 0; i < 4; ++i) atomic_add_f32(dp + i, sp[i]);
    if ((t & 31) == 0) atomic_add_f32(&cnt[d], 1.0f);
}

__global__ void seg_norm_kernel(float* __restrict__ agg, const float* __restrict__ cnt, int n) {
    size_t t = (size_t)blockIdx.x * blockDim.x + threadIdx.x;
    if (t >= (size_t)n * HD) return;
    agg[t] /= fmaxf(cnt[t >> 7], 1.0f);
}

// ---------- WMMA dual GEMM (in-place): agg <- agg @ Wl + x @ Wr ----------
// 32-row M tile, 4 waves; each wave holds 4 accumulators (2 row-groups x 2 n-tiles)
// and reuses each B fragment for both row-groups.
__global__ void sage_gemm_kernel(float* __restrict__ agg, const float* __restrict__ x,
                                 const unsigned short* __restrict__ Pl,
                                 const unsigned short* __restrict__ Pr, int nrows) {
    __shared__ unsigned short A_s[32 * APAD];
    __shared__ unsigned short X_s[32 * APAD];

    int tid  = threadIdx.x;
    int row0 = blockIdx.x * 32;

    // stage 32-row A / X tiles into LDS as bf16 (float4 reads, b64 packed stores)
    for (int i = tid; i < 32 * 32; i += 128) {
        int r = i >> 5, k = (i & 31) << 2;
        int gr = row0 + r;
        float4 av = make_float4(0.f, 0.f, 0.f, 0.f), xv = av;
        if (gr < nrows) {
            av = *(const float4*)&agg[(size_t)gr * HD + k];
            xv = *(const float4*)&x[(size_t)gr * HD + k];
        }
        union { unsigned short u[4]; uint2 d; } pa, px;
        pa.u[0] = f2bf(av.x); pa.u[1] = f2bf(av.y); pa.u[2] = f2bf(av.z); pa.u[3] = f2bf(av.w);
        px.u[0] = f2bf(xv.x); px.u[1] = f2bf(xv.y); px.u[2] = f2bf(xv.z); px.u[3] = f2bf(xv.w);
        *(uint2*)&A_s[r * APAD + k] = pa.d;
        *(uint2*)&X_s[r * APAD + k] = px.d;
    }
    __syncthreads();

    int lane = tid & 31;
    int w    = tid >> 5;          // wave id 0..3; wave covers cols [w*32, w*32+32), rows 0..31
    v8f c0 = {}, c1 = {}, c2 = {}, c3 = {};

#pragma unroll
    for (int kcIdx = 0; kcIdx < 4; ++kcIdx) {
        int kc = kcIdx << 5;
        v16bf a0 = load_frag_lds(A_s, 0,  lane, kc);
        v16bf a1 = load_frag_lds(A_s, 16, lane, kc);
        v16bf x0 = load_frag_lds(X_s, 0,  lane, kc);
        v16bf x1 = load_frag_lds(X_s, 16, lane, kc);
        v16bf b0l = load_frag_packed(Pl, lane, kcIdx, 2 * w);
        v16bf b0r = load_frag_packed(Pr, lane, kcIdx, 2 * w);
        v16bf b1l = load_frag_packed(Pl, lane, kcIdx, 2 * w + 1);
        v16bf b1r = load_frag_packed(Pr, lane, kcIdx, 2 * w + 1);
        c0 = __builtin_amdgcn_wmma_f32_16x16x32_bf16(false, a0, false, b0l, (short)0, c0, false, false);
        c0 = __builtin_amdgcn_wmma_f32_16x16x32_bf16(false, x0, false, b0r, (short)0, c0, false, false);
        c1 = __builtin_amdgcn_wmma_f32_16x16x32_bf16(false, a0, false, b1l, (short)0, c1, false, false);
        c1 = __builtin_amdgcn_wmma_f32_16x16x32_bf16(false, x0, false, b1r, (short)0, c1, false, false);
        c2 = __builtin_amdgcn_wmma_f32_16x16x32_bf16(false, a1, false, b0l, (short)0, c2, false, false);
        c2 = __builtin_amdgcn_wmma_f32_16x16x32_bf16(false, x1, false, b0r, (short)0, c2, false, false);
        c3 = __builtin_amdgcn_wmma_f32_16x16x32_bf16(false, a1, false, b1l, (short)0, c3, false, false);
        c3 = __builtin_amdgcn_wmma_f32_16x16x32_bf16(false, x1, false, b1r, (short)0, c3, false, false);
    }

    // C layout: VGPR r, lane l -> M = r + 8*(l>=16), N = l&15
    int mBase = row0 + ((lane & 16) ? 8 : 0);
    int n     = lane & 15;
#pragma unroll
    for (int r = 0; r < 8; ++r) {
        int rowA = mBase + r;          // row-group 0
        int rowB = rowA + 16;          // row-group 1
        if (rowA < nrows) {
            agg[(size_t)rowA * HD + w * 32 + n]      = c0[r];
            agg[(size_t)rowA * HD + w * 32 + 16 + n] = c1[r];
        }
        if (rowB < nrows) {
            agg[(size_t)rowB * HD + w * 32 + n]      = c2[r];
            agg[(size_t)rowB * HD + w * 32 + 16 + n] = c3[r];
        }
    }
}

// ---------- SiLU + LayerNorm + residual, one wave32 per row, float4 lanes ----------
__global__ void silu_ln_res_kernel(const float* __restrict__ nbuf, float* __restrict__ x,
                                   const float* __restrict__ b1, const float* __restrict__ b2,
                                   const float* __restrict__ g, const float* __restrict__ be,
                                   int nrows) {
    int row = blockIdx.x * 8 + (threadIdx.x >> 5);
    if (row >= nrows) return;
    int lane = threadIdx.x & 31;
    size_t base = (size_t)row * HD + lane * 4;
    int hb = lane * 4;
    float4 y  = *(const float4*)&nbuf[base];
    float4 a1 = *(const float4*)&b1[hb];
    float4 a2 = *(const float4*)&b2[hb];
    y.x = silu(y.x + a1.x + a2.x);
    y.y = silu(y.y + a1.y + a2.y);
    y.z = silu(y.z + a1.z + a2.z);
    y.w = silu(y.w + a1.w + a2.w);
    float s = y.x + y.y + y.z + y.w;
#pragma unroll
    for (int m = 16; m > 0; m >>= 1) s += __shfl_xor(s, m, 32);
    float mean = s * (1.0f / HD);
    float dx = y.x - mean, dy = y.y - mean, dz = y.z - mean, dw = y.w - mean;
    float v = dx * dx + dy * dy + dz * dz + dw * dw;
#pragma unroll
    for (int m = 16; m > 0; m >>= 1) v += __shfl_xor(v, m, 32);
    float rinv = rsqrtf(v * (1.0f / HD) + 1e-5f);
    float4 gg = *(const float4*)&g[hb];
    float4 bb = *(const float4*)&be[hb];
    float4 xo = *(const float4*)&x[base];
    float4 o;
    o.x = dx * rinv * gg.x + bb.x + xo.x;
    o.y = dy * rinv * gg.y + bb.y + xo.y;
    o.z = dz * rinv * gg.z + bb.z + xo.z;
    o.w = dw * rinv * gg.w + bb.w + xo.w;
    *(float4*)&x[base] = o;
}

// ---------- batch pooling ----------
__global__ void pool_accum_kernel(const float* __restrict__ x, const int* __restrict__ batch,
                                  float* __restrict__ sum, float* __restrict__ sq,
                                  float* __restrict__ cnt, int nrows) {
    size_t t = (size_t)blockIdx.x * blockDim.x + threadIdx.x;
    if (t >= (size_t)nrows * HD) return;
    int r = (int)(t >> 7), h = (int)(t & 127);
    int b = batch[r];
    float v = x[t];
    atomic_add_f32(&sum[b * HD + h], v);
    atomic_add_f32(&sq[b * HD + h], v * v);
    if (h == 0) atomic_add_f32(&cnt[b], 1.0f);
}

__global__ void pool_final_kernel(float* __restrict__ sum, float* __restrict__ sq,
                                  const float* __restrict__ cnt) {
    size_t t = (size_t)blockIdx.x * blockDim.x + threadIdx.x;
    if (t >= (size_t)NB * HD) return;
    float c = fmaxf(cnt[t >> 7], 1.0f);
    float m = sum[t] / c;
    float s = sq[t] / c;
    sum[t] = m;
    sq[t]  = sqrtf(fmaxf(s - m * m, 1e-6f));
}

// ---------- global feature branch ----------
__global__ void glob_kernel(const float* __restrict__ gf, const float* __restrict__ W,
                            const float* __restrict__ b, float* __restrict__ out) {
    int bb = blockIdx.x, h = threadIdx.x;
    float acc = b[h];
#pragma unroll
    for (int j = 0; j < GDIM; ++j) acc += gf[bb * GDIM + j] * W[j * HD + h];
    out[bb * HD + h] = silu(acc);
}

// ---------- backbone ----------
__global__ void backbone_kernel(const float* __restrict__ gm, const float* __restrict__ gstd,
                                const float* __restrict__ qm, const float* __restrict__ qstd,
                                const float* __restrict__ glob, const float* __restrict__ bk,
                                const float* __restrict__ pr, const float* __restrict__ W,
                                const float* __restrict__ b, float* __restrict__ feat) {
    int bb = blockIdx.x, h = threadIdx.x;
    float acc = b[h];
#pragma unroll 4
    for (int j = 0; j < HD; ++j) acc += gm[bb * HD + j]   * W[(0 * HD + j) * HD + h];
#pragma unroll 4
    for (int j = 0; j < HD; ++j) acc += gstd[bb * HD + j] * W[(1 * HD + j) * HD + h];
#pragma unroll 4
    for (int j = 0; j < HD; ++j) acc += qm[bb * HD + j]   * W[(2 * HD + j) * HD + h];
#pragma unroll 4
    for (int j = 0; j < HD; ++j) acc += qstd[bb * HD + j] * W[(3 * HD + j) * HD + h];
#pragma unroll 4
    for (int j = 0; j < HD; ++j) acc += glob[bb * HD + j] * W[(4 * HD + j) * HD + h];
    acc += bk[bb] * W[(5 * HD + 0) * HD + h];
    acc += pr[bb] * W[(5 * HD + 1) * HD + h];
    feat[bb * HD + h] = silu(acc);
}

// ---------- heads ----------
__global__ void heads_kernel(const float* __restrict__ feat, const float* __restrict__ thrW,
                             const float* __restrict__ thrb, const float* __restrict__ runW,
                             const float* __restrict__ runb, float* __restrict__ out) {
    int bb = blockIdx.x, t = threadIdx.x;
    if (t < 10) {
        float a = thrb[t];
#pragma unroll 4
        for (int h = 0; h < HD; ++h) a += feat[bb * HD + h] * thrW[h * 10 + t];
        out[bb * 10 + t] = a;
    } else if (t == 10) {
        float a = runb[0];
#pragma unroll 4
        for (int h = 0; h < HD; ++h) a += feat[bb * HD + h] * runW[h];
        out[NB * 10 + bb] = a;
    }
}

// ---------- host launch ----------
extern "C" void kernel_launch(void* const* d_in, const int* in_sizes, int n_in,
                              void* d_out, int out_size, void* d_ws, size_t ws_size,
                              hipStream_t stream) {
    const int*   gate_type   = (const int*)  d_in[0];
    const int*   gate_arity  = (const int*)  d_in[1];
    const int*   gate_isdir  = (const int*)  d_in[2];
    const float* gate_idxn   = (const float*)d_in[3];
    const float* qubit_deg   = (const float*)d_in[4];
    const int*   src_g2q     = (const int*)  d_in[5];
    const int*   dst_g2q     = (const int*)  d_in[6];
    const int*   src_q2g     = (const int*)  d_in[7];
    const int*   dst_q2g     = (const int*)  d_in[8];
    const int*   gate_batch  = (const int*)  d_in[9];
    const int*   qubit_batch = (const int*)  d_in[10];
    const float* global_feat = (const float*)d_in[11];
    const float* backend_bit = (const float*)d_in[12];
    const float* prec_bit    = (const float*)d_in[13];
    const float* emb_table   = (const float*)d_in[14];
    const float* gate_W      = (const float*)d_in[15];
    const float* gate_b      = (const float*)d_in[16];
    const float* qubit_W     = (const float*)d_in[17];
    const float* qubit_b     = (const float*)d_in[18];
    const float* g2q_ll_W    = (const float*)d_in[19];
    const float* g2q_ll_b    = (const float*)d_in[20];
    const float* g2q_lr_W    = (const float*)d_in[21];
    const float* g2q_lr_b    = (const float*)d_in[22];
    const float* q2g_ll_W    = (const float*)d_in[23];
    const float* q2g_ll_b    = (const float*)d_in[24];
    const float* q2g_lr_W    = (const float*)d_in[25];
    const float* q2g_lr_b    = (const float*)d_in[26];
    const float* gate_ln_g   = (const float*)d_in[27];
    const float* gate_ln_b   = (const float*)d_in[28];
    const float* qubit_ln_g  = (const float*)d_in[29];
    const float* qubit_ln_b  = (const float*)d_in[30];
    const float* glob_W      = (const float*)d_in[31];
    const float* glob_b      = (const float*)d_in[32];
    const float* bb_W        = (const float*)d_in[33];
    const float* bb_b        = (const float*)d_in[34];
    const float* thr_W       = (const float*)d_in[35];
    const float* thr_b       = (const float*)d_in[36];
    const float* run_W       = (const float*)d_in[37];
    const float* run_b       = (const float*)d_in[38];
    float* out = (float*)d_out;

    // workspace partition (floats; all block sizes keep 16B alignment)
    float* ws = (float*)d_ws;
    size_t o = 0;
    float* gx   = ws + o; o += (size_t)NG * HD;
    float* aggG = ws + o; o += (size_t)NG * HD;
    float* qx   = ws + o; o += (size_t)NQ * HD;
    float* aggQ = ws + o; o += (size_t)NQ * HD;
    float* cntG = ws + o; o += NG;
    float* cntQ = ws + o; o += NQ;
    float* gsum = ws + o; o += (size_t)NB * HD;
    float* gsq  = ws + o; o += (size_t)NB * HD;
    float* qsum = ws + o; o += (size_t)NB * HD;
    float* qsq  = ws + o; o += (size_t)NB * HD;
    float* gcnt = ws + o; o += NB;
    float* qcnt = ws + o; o += NB;
    float* glbf = ws + o; o += (size_t)NB * HD;
    float* feat = ws + o; o += (size_t)NB * HD;
    unsigned short* pw = (unsigned short*)(ws + o);      // 16 matrices * 16384 bf16 = 512KB
    unsigned short* p_g2q_ll = pw + 0 * NL * 16384;
    unsigned short* p_g2q_lr = pw + 1 * NL * 16384;
    unsigned short* p_q2g_ll = pw + 2 * NL * 16384;
    unsigned short* p_q2g_lr = pw + 3 * NL * 16384;

    // one-time weight packing into WMMA fragment-major bf16 (L2-resident)
    {
        unsigned grid = (unsigned)(((size_t)NL * 16384 + 255) / 256);
        pack_w_kernel<<<grid, 256, 0, stream>>>(g2q_ll_W, p_g2q_ll, NL);
        pack_w_kernel<<<grid, 256, 0, stream>>>(g2q_lr_W, p_g2q_lr, NL);
        pack_w_kernel<<<grid, 256, 0, stream>>>(q2g_ll_W, p_q2g_ll, NL);
        pack_w_kernel<<<grid, 256, 0, stream>>>(q2g_lr_W, p_q2g_lr, NL);
    }

    // input projections
    gate_proj_kernel<<<NG, HD, 0, stream>>>(gate_type, gate_arity, gate_isdir, gate_idxn,
                                            emb_table, gate_W, gate_b, gx, NG);
    qubit_proj_kernel<<<(unsigned)(((size_t)NQ * HD + 255) / 256), 256, 0, stream>>>(
        qubit_deg, qubit_W, qubit_b, qx, NQ);

    const unsigned scat_grid  = (unsigned)(((size_t)NE * 32 + 255) / 256);
    const unsigned normQ_grid = (unsigned)(((size_t)NQ * HD + 255) / 256);
    const unsigned normG_grid = (unsigned)(((size_t)NG * HD + 255) / 256);

    for (int l = 0; l < NL; ++l) {
        // agg_q = seg_mean(gx[src_g2q], dst_g2q)
        hipMemsetAsync(aggQ, 0, (size_t)NQ * HD * sizeof(float), stream);
        hipMemsetAsync(cntQ, 0, (size_t)NQ * sizeof(float), stream);
        scatter_add_kernel<<<scat_grid, 256, 0, stream>>>(gx, src_g2q, dst_g2q, aggQ, cntQ, NE);
        seg_norm_kernel<<<normQ_grid, 256, 0, stream>>>(aggQ, cntQ, NQ);
        // agg_g = seg_mean(qx[src_q2g], dst_q2g)   (uses pre-update qx)
        hipMemsetAsync(aggG, 0, (size_t)NG * HD * sizeof(float), stream);
        hipMemsetAsync(cntG, 0, (size_t)NG * sizeof(float), stream);
        scatter_add_kernel<<<scat_grid, 256, 0, stream>>>(qx, src_q2g, dst_q2g, aggG, cntG, NE);
        seg_norm_kernel<<<normG_grid, 256, 0, stream>>>(aggG, cntG, NG);

        // WMMA dual GEMMs (in-place over agg buffers), 32-row tiles
        sage_gemm_kernel<<<(NQ + 31) / 32, 128, 0, stream>>>(aggQ, qx, p_g2q_ll + l * 16384,
                                                             p_g2q_lr + l * 16384, NQ);
        sage_gemm_kernel<<<(NG + 31) / 32, 128, 0, stream>>>(aggG, gx, p_q2g_ll + l * 16384,
                                                             p_q2g_lr + l * 16384, NG);

        // residual updates (wave-per-row)
        silu_ln_res_kernel<<<(NG + 7) / 8, 256, 0, stream>>>(
            aggG, gx, q2g_ll_b + l * HD, q2g_lr_b + l * HD,
            gate_ln_g + l * HD, gate_ln_b + l * HD, NG);
        silu_ln_res_kernel<<<(NQ + 7) / 8, 256, 0, stream>>>(
            aggQ, qx, g2q_ll_b + l * HD, g2q_lr_b + l * HD,
            qubit_ln_g + l * HD, qubit_ln_b + l * HD, NQ);
    }

    // batch pooling
    hipMemsetAsync(gsum, 0, (size_t)NB * HD * sizeof(float), stream);
    hipMemsetAsync(gsq,  0, (size_t)NB * HD * sizeof(float), stream);
    hipMemsetAsync(qsum, 0, (size_t)NB * HD * sizeof(float), stream);
    hipMemsetAsync(qsq,  0, (size_t)NB * HD * sizeof(float), stream);
    hipMemsetAsync(gcnt, 0, (size_t)NB * sizeof(float), stream);
    hipMemsetAsync(qcnt, 0, (size_t)NB * sizeof(float), stream);
    pool_accum_kernel<<<(unsigned)(((size_t)NG * HD + 255) / 256), 256, 0, stream>>>(gx, gate_batch, gsum, gsq, gcnt, NG);
    pool_accum_kernel<<<(unsigned)(((size_t)NQ * HD + 255) / 256), 256, 0, stream>>>(qx, qubit_batch, qsum, qsq, qcnt, NQ);
    pool_final_kernel<<<(unsigned)(((size_t)NB * HD + 255) / 256), 256, 0, stream>>>(gsum, gsq, gcnt);
    pool_final_kernel<<<(unsigned)(((size_t)NB * HD + 255) / 256), 256, 0, stream>>>(qsum, qsq, qcnt);

    // global branch, backbone, heads
    glob_kernel<<<NB, HD, 0, stream>>>(global_feat, glob_W, glob_b, glbf);
    backbone_kernel<<<NB, HD, 0, stream>>>(gsum, gsq, qsum, qsq, glbf, backend_bit, prec_bit,
                                           bb_W, bb_b, feat);
    heads_kernel<<<NB, 32, 0, stream>>>(feat, thr_W, thr_b, run_W, run_b, out);
}
